// MultiHeadAttention_24988119728324
// MI455X (gfx1250) — matrix operand order repaired and checked by reading, not compiled, not run on hardware
//
#include <hip/hip_runtime.h>
#include <hip/hip_bf16.h>

// MHA forward for B=4, S=2048, D=1024, H=16 (DH=64) on gfx1250 (CDNA5, wave32).
// ~137 GFLOP dense f16 -> matrix-unit bound (memory ~250MB => ~11us at 23.3TB/s,
// far below WMMA time). All GEMMs via v_wmma_f32_16x16x32_f16; LDS staging uses
// GLOBAL_LOAD_ASYNC_TO_LDS (ASYNCcnt) + double-buffered tiles for overlap.

typedef __attribute__((ext_vector_type(16))) _Float16 v16h;
typedef __attribute__((ext_vector_type(8)))  _Float16 v8h;
typedef __attribute__((ext_vector_type(8)))  float    v8f;
typedef __attribute__((ext_vector_type(4)))  int      v4i;

constexpr int Bsz   = 4;
constexpr int Seq   = 2048;
constexpr int Dm    = 1024;
constexpr int Hn    = 16;
constexpr int Dh    = 64;
constexpr int Mrows = Bsz * Seq;            // 8192
constexpr float NEGM = -10000.0f;

#if __has_builtin(__builtin_amdgcn_global_load_async_to_lds_b128) && \
    __has_builtin(__builtin_amdgcn_s_wait_asynccnt)
#define HAS_ASYNC_LDS 1
#else
#define HAS_ASYNC_LDS 0
#endif

// 16-byte global -> LDS copy: async DMA path on gfx1250, manual fallback.
// Builtin params (from hipcc diagnostic): (v4i global*, v4i lds*, imm off, imm cpol).
__device__ __forceinline__ void stage16B(_Float16* lds, const _Float16* g) {
#if HAS_ASYNC_LDS
  __builtin_amdgcn_global_load_async_to_lds_b128(
      (__attribute__((address_space(1))) v4i*)
          (__attribute__((address_space(1))) void*)(void*)(g),
      (__attribute__((address_space(3))) v4i*)
          (__attribute__((address_space(3))) void*)(void*)(lds),
      0, 0);
#else
  *(v8h*)lds = *(const v8h*)g;
#endif
}
__device__ __forceinline__ void async_fence() {
#if HAS_ASYNC_LDS
  __builtin_amdgcn_s_wait_asynccnt(0);     // this wave's async->LDS complete
#endif
}

__device__ __forceinline__ v16h cat8(v8h lo, v8h hi8) {
  return __builtin_shufflevector(lo, hi8, 0,1,2,3,4,5,6,7,8,9,10,11,12,13,14,15);
}

// A-fragment (16x32 f16): lane m=l&15, khalf=l>>4.
// halves [0..7] = K(8*khalf + 0..7), [8..15] = K(16 + 8*khalf + 0..7).
__device__ __forceinline__ v16h load_a_frag(const _Float16* rowp, int hi) {
  v8h lo  = *(const v8h*)(rowp + 8*hi);
  v8h hi8 = *(const v8h*)(rowp + 16 + 8*hi);
  return cat8(lo, hi8);
}

// B-fragment (32x16 f16): lane n=l&15, khalf=l>>4.
// halves [0..7] = K(8*khalf + 0..7), [8..15] = K(16 + 8*khalf + 0..7).
__device__ __forceinline__ v16h load_b_frag(const _Float16* rowp, int hi) {
  v8h lo  = *(const v8h*)(rowp + 8*hi);
  v8h hi8 = *(const v8h*)(rowp + 16 + 8*hi);
  return cat8(lo, hi8);
}

__device__ __forceinline__ v8f wmma_f16(v16h a, v16h b, v8f c) {
  return __builtin_amdgcn_wmma_f32_16x16x32_f16(false, a, false, b, (short)0, c,
                                                false, false);
}

__device__ __forceinline__ void cvt16_f32_to_lds(_Float16* dst, const float* src) {
  const float4 f0 = ((const float4*)src)[0];
  const float4 f1 = ((const float4*)src)[1];
  const float4 f2 = ((const float4*)src)[2];
  const float4 f3 = ((const float4*)src)[3];
  v8h h0, h1;
  h0[0]=(_Float16)f0.x; h0[1]=(_Float16)f0.y; h0[2]=(_Float16)f0.z; h0[3]=(_Float16)f0.w;
  h0[4]=(_Float16)f1.x; h0[5]=(_Float16)f1.y; h0[6]=(_Float16)f1.z; h0[7]=(_Float16)f1.w;
  h1[0]=(_Float16)f2.x; h1[1]=(_Float16)f2.y; h1[2]=(_Float16)f2.z; h1[3]=(_Float16)f2.w;
  h1[4]=(_Float16)f3.x; h1[5]=(_Float16)f3.y; h1[6]=(_Float16)f3.z; h1[7]=(_Float16)f3.w;
  *(v8h*)dst       = h0;
  *(v8h*)(dst + 8) = h1;
}

// ---------------------------------------------------------------------------
// GEMM: Y = X @ W^T + bias.  M=8192, N=K=1024.  WG tile 128x128, BK=32,
// double-buffered LDS: one barrier/iter, next tile staged during WMMAs.
// 8 waves (wave32): wave grid 4(M) x 2(N), each wave 32x64 = 2x4 WMMA tiles.
// ---------------------------------------------------------------------------
template<bool A_F16, bool OUT_HEAD>
__global__ __launch_bounds__(256)
void gemm_xWt_kernel(const void* __restrict__ Ap,
                     const float* __restrict__ W,
                     const float* __restrict__ bias,
                     void* __restrict__ Out)
{
  constexpr int BM = 128, BN = 128, BK = 32, PK = BK + 8;  // 80B rows: 16B-aligned, conflict-free
  __shared__ alignas(16) _Float16 As[2][BM][PK];
  __shared__ alignas(16) _Float16 Bs[2][BN][PK];

  const int t    = threadIdx.x;
  const int lane = t & 31;
  const int wv   = t >> 5;
  const int hi   = lane >> 4;
  const int ln   = lane & 15;
  const int wm   = (wv >> 1) * 32;
  const int wn   = (wv & 1) * 64;
  const int m0   = blockIdx.x * BM;
  const int n0   = blockIdx.y * BN;
  const int lrow = t >> 1;           // 0..127
  const int lk   = (t & 1) * 16;     // 0 or 16

  auto stage = [&](int p, int k0) {
    if constexpr (A_F16) {
      const _Float16* a16 = (const _Float16*)Ap + (size_t)(m0 + lrow) * Dm + k0 + lk;
      stage16B(&As[p][lrow][lk],     a16);       // async -> LDS
      stage16B(&As[p][lrow][lk + 8], a16 + 8);
    } else {
      const float* a32 = (const float*)Ap + (size_t)(m0 + lrow) * Dm + k0 + lk;
      cvt16_f32_to_lds(&As[p][lrow][lk], a32);   // f32->f16 convert in VALU
    }
    // B[k][n] = W[n][k]: stage W rows directly; B-frag gather transposes.
    cvt16_f32_to_lds(&Bs[p][lrow][lk], W + (size_t)(n0 + lrow) * Dm + k0 + lk);
  };

  v8f acc[2][4] = {};
  constexpr int nIter = Dm / BK;     // 32

  stage(0, 0);
  for (int it = 0; it < nIter; ++it) {
    const int p = it & 1;
    async_fence();                   // buf p async writes landed (this wave)
    __syncthreads();                 // + all waves' ds writes / prior reads done
    if (it + 1 < nIter) stage(1 - p, (it + 1) * BK);   // overlap with WMMAs

    v16h afr[2], bfr[4];
#pragma unroll
    for (int mf = 0; mf < 2; ++mf) afr[mf] = load_a_frag(&As[p][wm + mf*16 + ln][0], hi);
#pragma unroll
    for (int nf = 0; nf < 4; ++nf) bfr[nf] = load_b_frag(&Bs[p][wn + nf*16 + ln][0], hi);
#pragma unroll
    for (int mf = 0; mf < 2; ++mf)
#pragma unroll
      for (int nf = 0; nf < 4; ++nf)
        acc[mf][nf] = wmma_f16(afr[mf], bfr[nf], acc[mf][nf]);
  }

  // C/D layout: VGPR r -> (m = r + 8*hi, n = ln)
#pragma unroll
  for (int nf = 0; nf < 4; ++nf) {
    const int n  = n0 + wn + nf*16 + ln;
    const float bv = bias[n];
#pragma unroll
    for (int mf = 0; mf < 2; ++mf) {
#pragma unroll
      for (int r = 0; r < 8; ++r) {
        const int m = m0 + wm + mf*16 + r + 8*hi;
        const float val = acc[mf][nf][r] + bv;
        if constexpr (OUT_HEAD) {
          const int hh = n >> 6, dh = n & (Dh - 1);
          const int bb = m >> 11, ss = m & (Seq - 1);
          ((_Float16*)Out)[(((size_t)bb * Hn + hh) * Seq + ss) * Dh + dh] = (_Float16)val;
        } else {
          ((float*)Out)[(size_t)m * Dm + n] = val;
        }
      }
    }
  }
}

// ---------------------------------------------------------------------------
// Flash attention per (b,h): causal, scale 1/8. Q tile 128 rows, K/V tiles 128,
// double-buffered with async K staging. 8 waves; wave w owns rows [16w,16w+16):
// row stats wave-private, P LDS round-trip needs no barrier (same-wave dscnt).
// ---------------------------------------------------------------------------
__global__ __launch_bounds__(256)
void flash_attn_kernel(const _Float16* __restrict__ Qh,
                       const _Float16* __restrict__ Kh,
                       const _Float16* __restrict__ Vh,
                       _Float16* __restrict__ AttnOut)
{
  __shared__ alignas(16) _Float16 Qs[128][72];       // [q][dh]   144B rows
  __shared__ alignas(16) _Float16 Ks[2][128][72];    // [k][dh]   double-buffered
  __shared__ alignas(16) _Float16 Vt[2][64][136];    // [dh][k]   272B rows (transposed)
  __shared__ alignas(16) _Float16 Ps[128][136];      // [q][k]    P = exp(S - m)

  const int t    = threadIdx.x;
  const int lane = t & 31;
  const int wv   = t >> 5;
  const int hi   = lane >> 4;
  const int ln   = lane & 15;
  const int qt   = blockIdx.x;           // 0..15
  const int bh   = blockIdx.y;           // 0..63
  const _Float16* Qb = Qh + (size_t)bh * Seq * Dh;
  const _Float16* Kb = Kh + (size_t)bh * Seq * Dh;
  const _Float16* Vb = Vh + (size_t)bh * Seq * Dh;
  const int q0  = qt * 128;
  const int row = t >> 1;                // 0..127
  const int seg = (t & 1) * 32;          // 0 or 32 halves

  {
    const _Float16* qp = Qb + (size_t)(q0 + row) * Dh + seg;
    stage16B(&Qs[row][seg],      qp);
    stage16B(&Qs[row][seg + 8],  qp + 8);
    stage16B(&Qs[row][seg + 16], qp + 16);
    stage16B(&Qs[row][seg + 24], qp + 24);
  }
  async_fence();
  __syncthreads();
  v16h aq[2];
  aq[0] = load_a_frag(&Qs[wv*16 + ln][0],  hi);
  aq[1] = load_a_frag(&Qs[wv*16 + ln][32], hi);

  auto stageKV = [&](int p, int ktile) {
    const int kb = ktile * 128;
    const _Float16* kp = Kb + (size_t)(kb + row) * Dh + seg;
    stage16B(&Ks[p][row][seg],      kp);          // async -> LDS
    stage16B(&Ks[p][row][seg + 8],  kp + 8);
    stage16B(&Ks[p][row][seg + 16], kp + 16);
    stage16B(&Ks[p][row][seg + 24], kp + 24);
    const _Float16* vp = Vb + (size_t)(kb + row) * Dh + seg;
    v8h a0 = *(const v8h*)(vp);
    v8h a1 = *(const v8h*)(vp + 8);
    v8h a2 = *(const v8h*)(vp + 16);
    v8h a3 = *(const v8h*)(vp + 24);
#pragma unroll
    for (int j = 0; j < 8; ++j) {                 // transpose V into Vt[dh][k]
      Vt[p][seg + j     ][row] = a0[j];
      Vt[p][seg + j + 8 ][row] = a1[j];
      Vt[p][seg + j + 16][row] = a2[j];
      Vt[p][seg + j + 24][row] = a3[j];
    }
  };

  v8f oacc[4] = {};
  float mi[8], li[8];
#pragma unroll
  for (int r = 0; r < 8; ++r) { mi[r] = -3.0e38f; li[r] = 0.0f; }

  stageKV(0, 0);
  for (int kt = 0; kt <= qt; ++kt) {          // causal: only tiles kt <= qt
    const int p = kt & 1;
    async_fence();                             // buf p K-tile landed (this wave)
    __syncthreads();                           // all waves: buf p ready, prior reads done
    if (kt + 1 <= qt) stageKV(1 - p, kt + 1);  // overlap next tile with compute
    if (kt + 2 <= qt) {                        // L2 prefetch two tiles ahead
      __builtin_prefetch(Kb + (size_t)((kt + 2) * 128 + row) * Dh + seg, 0, 3);
      __builtin_prefetch(Vb + (size_t)((kt + 2) * 128 + row) * Dh + seg, 0, 3);
    }

    // S = Q @ K^T  (B[k=dh][n=sk] = K[sk][dh] -> read Ks rows)
    v8f sacc[8] = {};
#pragma unroll
    for (int ks = 0; ks < 2; ++ks) {
#pragma unroll
      for (int nf = 0; nf < 8; ++nf) {
        v16h bfr = load_b_frag(&Ks[p][nf*16 + ln][ks*32], hi);
        sacc[nf] = wmma_f16(aq[ks], bfr, sacc[nf]);
      }
    }

    const bool diag = (kt == qt);
#pragma unroll
    for (int nf = 0; nf < 8; ++nf) {
#pragma unroll
      for (int r = 0; r < 8; ++r) {
        float s = sacc[nf][r] * 0.125f;        // 1/sqrt(64)
        if (diag) {
          const int nk = nf*16 + ln;
          const int ml = wv*16 + r + 8*hi;
          s += (nk > ml) ? NEGM : 0.0f;        // causal mask on diagonal tile
        }
        sacc[nf][r] = s;
      }
    }

    // Row max across the 16-lane half-wave holding this row's columns.
    float sc[8];
#pragma unroll
    for (int r = 0; r < 8; ++r) {
      float rm = sacc[0][r];
#pragma unroll
      for (int nf = 1; nf < 8; ++nf) rm = fmaxf(rm, sacc[nf][r]);
      rm = fmaxf(rm, __shfl_xor(rm, 1));
      rm = fmaxf(rm, __shfl_xor(rm, 2));
      rm = fmaxf(rm, __shfl_xor(rm, 4));
      rm = fmaxf(rm, __shfl_xor(rm, 8));
      const float mn = fmaxf(mi[r], rm);
      sc[r] = __expf(mi[r] - mn);
      mi[r] = mn;
    }

#pragma unroll
    for (int r = 0; r < 8; ++r) {
      float rs = 0.0f;
#pragma unroll
      for (int nf = 0; nf < 8; ++nf) {
        const float pe = __expf(sacc[nf][r] - mi[r]);
        sacc[nf][r] = pe;
        rs += pe;
        Ps[wv*16 + r + 8*hi][nf*16 + ln] = (_Float16)pe;  // C/D -> A layout via LDS
      }
      rs += __shfl_xor(rs, 1);
      rs += __shfl_xor(rs, 2);
      rs += __shfl_xor(rs, 4);
      rs += __shfl_xor(rs, 8);
      li[r] = li[r] * sc[r] + rs;
    }

#pragma unroll
    for (int no = 0; no < 4; ++no)
#pragma unroll
      for (int r = 0; r < 8; ++r)
        oacc[no][r] *= sc[r];

    // O += P @ V  (B[k=sk][n=dh] = Vt[dh][sk] -> read Vt rows)
#pragma unroll
    for (int ks2 = 0; ks2 < 4; ++ks2) {
      v16h pa = load_a_frag(&Ps[wv*16 + ln][ks2*32], hi);
#pragma unroll
      for (int no = 0; no < 4; ++no) {
        v16h vb = load_b_frag(&Vt[p][no*16 + ln][ks2*32], hi);
        oacc[no] = wmma_f16(pa, vb, oacc[no]);
      }
    }
  }

  const int bb = bh >> 4, hh = bh & 15;
#pragma unroll
  for (int no = 0; no < 4; ++no) {
#pragma unroll
    for (int r = 0; r < 8; ++r) {
      const float o = oacc[no][r] / li[r];
      const int s = q0 + wv*16 + r + 8*hi;
      const int n = hh*Dh + no*16 + ln;
      AttnOut[((size_t)bb * Seq + s) * Dm + n] = (_Float16)o;  // [B,S,D] f16
    }
  }
}

// ---------------------------------------------------------------------------
extern "C" void kernel_launch(void* const* d_in, const int* in_sizes, int n_in,
                              void* d_out, int out_size, void* d_ws, size_t ws_size,
                              hipStream_t stream) {
  (void)in_sizes; (void)n_in; (void)out_size; (void)ws_size;
  const float* q  = (const float*)d_in[0];
  const float* k  = (const float*)d_in[1];
  const float* v  = (const float*)d_in[2];
  /* d_in[3] = mask: causal triu, applied analytically */
  const float* Wq = (const float*)d_in[4];
  const float* bq = (const float*)d_in[5];
  const float* Wk = (const float*)d_in[6];
  const float* bk = (const float*)d_in[7];
  const float* Wv = (const float*)d_in[8];
  const float* bv = (const float*)d_in[9];
  const float* Wo = (const float*)d_in[10];
  const float* bo = (const float*)d_in[11];
  float* out = (float*)d_out;

  const size_t BSD = (size_t)Bsz * Seq * Dm;   // 8,388,608 elems (f16) per buffer
  _Float16* Qh   = (_Float16*)d_ws;            // [B,H,S,DH]
  _Float16* Kh   = Qh + BSD;
  _Float16* Vh   = Kh + BSD;
  _Float16* Attn = Vh + BSD;                   // [B,S,D]  (64 MB workspace total)

  dim3 blk(256);
  dim3 gp(Mrows / 128, Dm / 128);              // 64 x 8
  gemm_xWt_kernel<false, true><<<gp, blk, 0, stream>>>(q, Wq, bq, Qh);
  gemm_xWt_kernel<false, true><<<gp, blk, 0, stream>>>(k, Wk, bk, Kh);
  gemm_xWt_kernel<false, true><<<gp, blk, 0, stream>>>(v, Wv, bv, Vh);

  dim3 ga(Seq / 128, Bsz * Hn);                // 16 x 64
  flash_attn_kernel<<<ga, blk, 0, stream>>>(Qh, Kh, Vh, Attn);

  gemm_xWt_kernel<true, false><<<gp, blk, 0, stream>>>(Attn, Wo, bo, out);
}
